// HAT_RD_3229815407004
// MI455X (gfx1250) — compile-verified, matrix operand order checked by reading, not run on hardware
//
#include <hip/hip_runtime.h>
#include <hip/hip_bf16.h>
#include <math.h>

// ---------------------------------------------------------------------------
// HAT / hierarchical BiLSTM on MI455X (gfx1250, wave32, WMMA bf16).
//
//   1. init_padded + build_padded : ragged token columns -> [16][256][64]
//   2. emb_cast   : emb f32 [30000,300] -> bf16 [30000,320]
//   3. tcast x8   : weights f32 [K,N] -> bf16 [N,Kp] transposed+padded
//   4. word_lstm  : 64 rows/WG x 2 dirs; async-LDS x-staging; B reused over
//                   4 M-tiles; in-register i/f/g/o cell update; h double-buf
//   5. mask_fc    : post-validity mask -> 1.0, word FC, bf16 re-store
//   6. post_xgemm : parallel WMMA GEMM of the non-recurrent input projection
//   7. post_lstm  : serial 256-step recurrence, Whh only, in-register gates
//   8. final_fc   : [16,1024] . [1024] -> d_out[0..15]
// Output: d_out = [ result (16) | word_result (4096) ]  (float32)
// ---------------------------------------------------------------------------

typedef __attribute__((ext_vector_type(16))) __bf16 v16bf;
typedef __attribute__((ext_vector_type(8)))  float  v8f;

#define SEPTOK 2
#define SLEN   4096
#define BATCH  16
#define PMAX   256
#define WMAXW  64
#define EW     300
#define EWP    320
#define HW     256      // word hidden
#define GW     1024     // 4*HW
#define HP     512      // post hidden
#define GP     2048     // 4*HP
#define NROWS  4096
#define VOCAB  30000

__device__ __forceinline__ float rcp_f(float x) { return __builtin_amdgcn_rcpf(x); }
__device__ __forceinline__ float sig_f(float x) { return rcp_f(1.0f + __expf(-x)); }
__device__ __forceinline__ float tanh_f(float x) {
    return 1.0f - 2.0f * rcp_f(__expf(2.0f * x) + 1.0f);
}

// One lane's slice of a 16x32 bf16 fragment: row-major source row, lane holds
// K = k0+kb..+7 and k0+kb+16..+23, kb = (lane>=16)*8 (ISA 16-bit A layout).
__device__ __forceinline__ v16bf load_frag(const __bf16* __restrict__ row, int k0, int kb) {
    union { uint4 u[2]; v16bf v; } u;
    u.u[0] = *(const uint4*)(row + k0 + kb);
    u.u[1] = *(const uint4*)(row + k0 + kb + 16);
    return u.v;
}

__device__ __forceinline__ v8f wmma_bf16(v16bf a, v16bf b, v8f c) {
    return __builtin_amdgcn_wmma_f32_16x16x32_bf16(false, a, false, b, (short)0, c,
                                                   false, false);
}

// Async copy 16 bytes global -> LDS (gfx1250 ASYNCcnt path).
__device__ __forceinline__ void async_b128(unsigned lds_off, const void* gaddr) {
    asm volatile("global_load_async_to_lds_b128 %0, %1, off"
                 :: "v"(lds_off), "v"(gaddr) : "memory");
}
__device__ __forceinline__ void wait_async0() {
    asm volatile("s_wait_asynccnt 0x0" ::: "memory");
}

// ---------------------------------------------------------------------------
__global__ __launch_bounds__(256) void init_padded_kernel(int* __restrict__ padded) {
    int idx = blockIdx.x * 256 + threadIdx.x;
    if (idx < BATCH * PMAX * WMAXW) padded[idx] = 1;
}

__global__ __launch_bounds__(256) void build_padded_kernel(
    const int* __restrict__ batch, int* __restrict__ padded, int* __restrict__ n_posts) {
    const int b = blockIdx.x;
    const int tid = threadIdx.x;
    __shared__ int s_cnt[256];
    __shared__ int s_last[256];
    const int CH = SLEN / 256;
    const int base = tid * CH;
    int cnt = 0, last = -1;
    for (int i = 0; i < CH; ++i) {
        int pos = base + i;
        int tk = batch[pos * BATCH + b];
        if (tk == SEPTOK) { cnt++; last = pos; }
    }
    s_cnt[tid] = cnt; s_last[tid] = last;
    __syncthreads();
    for (int off = 1; off < 256; off <<= 1) {
        int ac = (tid >= off) ? s_cnt[tid - off] : 0;
        int al = (tid >= off) ? s_last[tid - off] : -1;
        __syncthreads();
        s_cnt[tid] += ac;
        s_last[tid] = max(s_last[tid], al);
        __syncthreads();
    }
    int post = (tid > 0) ? s_cnt[tid - 1] : 0;
    int lastsep = (tid > 0) ? s_last[tid - 1] : -1;
    for (int i = 0; i < CH; ++i) {
        int pos = base + i;
        int tk = batch[pos * BATCH + b];
        if (tk == SEPTOK) { post++; lastsep = pos; }
        else {
            int w = pos - lastsep - 1;
            if (w < WMAXW && post < PMAX)
                padded[((b << 8) + post) * WMAXW + w] = tk;
        }
    }
    if (tid == 255) n_posts[b] = min(s_cnt[255], PMAX);
}

// ---------------------------------------------------------------------------
__global__ __launch_bounds__(256) void emb_cast_kernel(
    const float* __restrict__ emb, __bf16* __restrict__ out) {
    long long idx = (long long)blockIdx.x * 256 + threadIdx.x;
    if (idx >= (long long)VOCAB * EWP) return;
    int v = (int)(idx / EWP), e = (int)(idx % EWP);
    out[idx] = (__bf16)((e < EW) ? emb[(long long)v * EW + e] : 0.0f);
}

__global__ __launch_bounds__(256) void tcast_kernel(
    const float* __restrict__ in, __bf16* __restrict__ out, int K, int N, int Kp) {
    long long idx = (long long)blockIdx.x * 256 + threadIdx.x;
    if (idx >= (long long)N * Kp) return;
    int n = (int)(idx / Kp), k = (int)(idx % Kp);
    out[idx] = (__bf16)((k < K) ? in[(long long)k * N + n] : 0.0f);
}

// ---------------------------------------------------------------------------
// 4. Word BiLSTM. grid = (64 row-blocks, 2 dirs), block = 256 (8 waves).
//    64 rows (4 M-tiles) per WG; wave w owns hidden cols [w*32, w*32+32)
//    across all 4 gates. x_t staged to LDS via async loads; B tiles reused
//    across the 4 M-tiles; cell update fully in registers.
//    LDS: Xs bf16 [64][320] | H bf16 2x[64][256] | C f32 [64][256] = 168 KB.
__global__ __launch_bounds__(256) void word_lstm_kernel(
    const int* __restrict__ padded,          // [4096][64]
    const __bf16* __restrict__ emb16,        // [30000][320]
    const __bf16* __restrict__ wihT,         // [2][1024][320]
    const __bf16* __restrict__ whhT,         // [2][1024][256]
    const float* __restrict__ wb_f, const float* __restrict__ wb_b,
    float* __restrict__ wordh)               // [4096][512]
{
    const int nbase = blockIdx.x * 64;
    const int dir   = blockIdx.y;
    const __bf16* WI = wihT + (size_t)dir * GW * EWP;
    const __bf16* WH = whhT + (size_t)dir * GW * HW;
    const float* bias = dir ? wb_b : wb_f;

    extern __shared__ char smem[];
    __bf16* Xs = (__bf16*)smem;                       // [64][320]
    __bf16* H0 = Xs + 64 * EWP;                       // [64][256]
    __bf16* H1 = H0 + 64 * HW;                        // [64][256]
    float*  Cld = (float*)(H1 + 64 * HW);             // [64][256]
    const unsigned xs_base = (unsigned)(size_t)Xs;    // LDS byte address

    const int tid = threadIdx.x;
    const int lane = tid & 31;
    const int wave = tid >> 5;
    const int lm = lane & 15;
    const int kb = (lane >> 4) << 3;
    const int mo = (lane >> 4) << 3;

    for (int e = tid; e < 64 * HW; e += 256) { Cld[e] = 0.0f; H0[e] = (__bf16)0.0f; }
    __syncthreads();

    const int srow = tid >> 2;            // staging: row 0..63
    const int sq   = tid & 3;             // quarter of the row (80 elems)
    const int* srow_toks = padded + (size_t)(nbase + srow) * WMAXW;

    for (int step = 0; step < WMAXW; ++step) {
        const int t = dir ? (WMAXW - 1 - step) : step;
        // ---- stage x_t rows into LDS (async) ----
        {
            const int tok = srow_toks[t];
            const __bf16* src = emb16 + (size_t)tok * EWP + sq * 80;
            unsigned dst = xs_base + (unsigned)(srow * EWP + sq * 80) * 2u;
            #pragma unroll
            for (int c = 0; c < 10; ++c)
                async_b128(dst + c * 16u, (const void*)(src + c * 8));
            if (step + 1 < WMAXW) {
                int tn = srow_toks[dir ? (WMAXW - 2 - step) : (step + 1)];
                __builtin_prefetch((const void*)(emb16 + (size_t)tn * EWP), 0, 1);
            }
            wait_async0();
        }
        __syncthreads();

        const __bf16* Hr = (step & 1) ? H1 : H0;
        __bf16*       Hw = (step & 1) ? H0 : H1;

        #pragma unroll
        for (int sub = 0; sub < 2; ++sub) {
            const int colh = wave * 32 + sub * 16 + lm;     // hidden col 0..255
            v8f acc[4][4];                                  // [mtile][gate]
            #pragma unroll
            for (int g = 0; g < 4; ++g) {
                const float bv = bias[g * HW + colh];
                #pragma unroll
                for (int m = 0; m < 4; ++m)
                    #pragma unroll
                    for (int r = 0; r < 8; ++r) acc[m][g][r] = bv;
            }
            const __bf16* wrow[4];
            #pragma unroll
            for (int g = 0; g < 4; ++g) wrow[g] = WI + (size_t)(g * HW + colh) * EWP;

            for (int kk = 0; kk < 10; ++kk) {               // x part: K = 320
                v16bf A[4];
                #pragma unroll
                for (int m = 0; m < 4; ++m)
                    A[m] = load_frag(Xs + (m * 16 + lm) * EWP, kk * 32, kb);
                #pragma unroll
                for (int g = 0; g < 4; ++g) {
                    v16bf B = load_frag(wrow[g], kk * 32, kb);
                    #pragma unroll
                    for (int m = 0; m < 4; ++m)
                        acc[m][g] = wmma_bf16(A[m], B, acc[m][g]);
                }
            }
            #pragma unroll
            for (int g = 0; g < 4; ++g) wrow[g] = WH + (size_t)(g * HW + colh) * HW;

            for (int kk = 0; kk < 8; ++kk) {                // h part: K = 256
                v16bf A[4];
                #pragma unroll
                for (int m = 0; m < 4; ++m)
                    A[m] = load_frag(Hr + (m * 16 + lm) * HW, kk * 32, kb);
                #pragma unroll
                for (int g = 0; g < 4; ++g) {
                    v16bf B = load_frag(wrow[g], kk * 32, kb);
                    #pragma unroll
                    for (int m = 0; m < 4; ++m)
                        acc[m][g] = wmma_bf16(A[m], B, acc[m][g]);
                }
            }
            // ---- in-register LSTM cell update ----
            #pragma unroll
            for (int m = 0; m < 4; ++m)
                #pragma unroll
                for (int r = 0; r < 8; ++r) {
                    const int row = m * 16 + r + mo;
                    const int idx = row * HW + colh;
                    float gi = acc[m][0][r], gf = acc[m][1][r];
                    float gg = acc[m][2][r], go = acc[m][3][r];
                    float c = sig_f(gf) * Cld[idx] + sig_f(gi) * tanh_f(gg);
                    Cld[idx] = c;
                    Hw[idx] = (__bf16)(sig_f(go) * tanh_f(c));
                }
        }
        __syncthreads();
    }
    // final h lives in H0 (64 steps, even number of swaps)
    for (int e = tid; e < 64 * HW; e += 256) {
        const int row = e >> 8, hcol = e & 255;
        wordh[(size_t)(nbase + row) * (2 * HW) + dir * HW + hcol] = (float)H0[e];
    }
}

// ---------------------------------------------------------------------------
// 5. mask invalid posts to 1.0, word FC, bf16 restore. grid 4096, block 32.
__global__ __launch_bounds__(32) void mask_fc_kernel(
    const float* __restrict__ wordh, const int* __restrict__ n_posts,
    const float* __restrict__ fcw, const float* __restrict__ fcb,
    __bf16* __restrict__ wordh16, float* __restrict__ out_wordres) {
    const int row = blockIdx.x;                 // b*256 + p
    const int b = row >> 8, p = row & 255;
    const bool valid = p < n_posts[b];
    const int lane = threadIdx.x;
    float sum = 0.0f;
    for (int e = lane; e < 2 * HW; e += 32) {
        float v = valid ? wordh[(size_t)row * (2 * HW) + e] : 1.0f;
        wordh16[(size_t)row * (2 * HW) + e] = (__bf16)v;
        sum += v * fcw[e];
    }
    for (int off = 16; off > 0; off >>= 1) sum += __shfl_down(sum, off, 32);
    if (lane == 0) out_wordres[row] = sum + fcb[0];
}

// ---------------------------------------------------------------------------
// 6. Non-recurrent input projection of the post LSTM, fully parallel:
//    Gx[dir][t][b][2048] = x(t,b) @ pihT + bias.  grid (256, 2), block 256.
__global__ __launch_bounds__(256) void post_xgemm_kernel(
    const __bf16* __restrict__ wordh16,      // [16*256][512]
    const __bf16* __restrict__ pihT,         // [2][2048][512]
    const float* __restrict__ pb_f, const float* __restrict__ pb_b,
    float* __restrict__ Gx)                  // [2][256][16][2048]
{
    const int t   = blockIdx.x;
    const int dir = blockIdx.y;
    const __bf16* WI = pihT + (size_t)dir * GP * HP;
    const float* bias = dir ? pb_b : pb_f;

    const int tid = threadIdx.x;
    const int lane = tid & 31;
    const int wave = tid >> 5;
    const int lm = lane & 15;
    const int kb = (lane >> 4) << 3;
    const int mo = (lane >> 4) << 3;

    const __bf16* xr = wordh16 + ((size_t)lm * PMAX + t) * HP;   // batch row lm
    v16bf Ax[16];
    #pragma unroll
    for (int kk = 0; kk < 16; ++kk) Ax[kk] = load_frag(xr, kk * 32, kb);

    float* gout = Gx + (size_t)(dir * PMAX + t) * 16 * GP;
    for (int j = 0; j < 16; ++j) {
        const int col = wave * 256 + j * 16 + lm;
        const float bv = bias[col];
        v8f acc;
        #pragma unroll
        for (int r = 0; r < 8; ++r) acc[r] = bv;
        const __bf16* wi = WI + (size_t)col * HP;
        #pragma unroll
        for (int kk = 0; kk < 16; ++kk) {
            v16bf B = load_frag(wi, kk * 32, kb);
            acc = wmma_bf16(Ax[kk], B, acc);
        }
        #pragma unroll
        for (int r = 0; r < 8; ++r) gout[(size_t)(r + mo) * GP + col] = acc[r];
    }
}

// ---------------------------------------------------------------------------
// 7. Post recurrence: only h @ Whh per step (Gx precomputed). grid (2),
//    block 256: wave w owns hidden cols [w*64, w*64+64) across all 4 gates,
//    in-register cell update, double-buffered h.
//    LDS: H bf16 2x[16][512] | C f32 [16][512] = 64 KB.
__global__ __launch_bounds__(256) void post_lstm_kernel(
    const float* __restrict__ Gx,            // [2][256][16][2048]
    const __bf16* __restrict__ phhT,         // [2][2048][512]
    float* __restrict__ posth)               // [16][1024]
{
    const int dir = blockIdx.x;
    const __bf16* WH = phhT + (size_t)dir * GP * HP;

    extern __shared__ char smem[];
    __bf16* H0 = (__bf16*)smem;                      // [16][512]
    __bf16* H1 = H0 + 16 * HP;                       // [16][512]
    float*  Cld = (float*)(H1 + 16 * HP);            // [16][512]

    const int tid = threadIdx.x;
    const int lane = tid & 31;
    const int wave = tid >> 5;
    const int lm = lane & 15;
    const int kb = (lane >> 4) << 3;
    const int mo = (lane >> 4) << 3;

    for (int e = tid; e < 16 * HP; e += 256) { Cld[e] = 0.0f; H0[e] = (__bf16)0.0f; }
    __syncthreads();

    for (int step = 0; step < PMAX; ++step) {
        const int t = dir ? (PMAX - 1 - step) : step;
        const __bf16* Hr = (step & 1) ? H1 : H0;
        __bf16*       Hw = (step & 1) ? H0 : H1;
        const float* gx = Gx + (size_t)(dir * PMAX + t) * 16 * GP;

        #pragma unroll
        for (int sub = 0; sub < 4; ++sub) {
            const int colh = wave * 64 + sub * 16 + lm;     // hidden col 0..511
            v8f acc[4];                                     // per gate
            #pragma unroll
            for (int g = 0; g < 4; ++g)
                #pragma unroll
                for (int r = 0; r < 8; ++r)
                    acc[g][r] = gx[(size_t)(r + mo) * GP + g * HP + colh];

            const __bf16* hrow = Hr + lm * HP;
            for (int kk = 0; kk < 16; ++kk) {               // K = 512 (h part)
                v16bf A = load_frag(hrow, kk * 32, kb);
                #pragma unroll
                for (int g = 0; g < 4; ++g) {
                    v16bf B = load_frag(WH + (size_t)(g * HP + colh) * HP, kk * 32, kb);
                    acc[g] = wmma_bf16(A, B, acc[g]);
                }
            }
            #pragma unroll
            for (int r = 0; r < 8; ++r) {
                const int row = r + mo;
                const int idx = row * HP + colh;
                float gi = acc[0][r], gf = acc[1][r], gg = acc[2][r], go = acc[3][r];
                float c = sig_f(gf) * Cld[idx] + sig_f(gi) * tanh_f(gg);
                Cld[idx] = c;
                Hw[idx] = (__bf16)(sig_f(go) * tanh_f(c));
            }
        }
        __syncthreads();
    }
    // final h lives in H0 (256 steps, even)
    for (int e = tid; e < 16 * HP; e += 256) {
        const int row = e >> 9, hcol = e & 511;
        posth[(size_t)row * (2 * HP) + dir * HP + hcol] = (float)H0[e];
    }
}

// ---------------------------------------------------------------------------
__global__ __launch_bounds__(32) void final_fc_kernel(
    const float* __restrict__ posth, const float* __restrict__ fcw,
    const float* __restrict__ fcb, float* __restrict__ out) {
    const int b = blockIdx.x, lane = threadIdx.x;
    float sum = 0.0f;
    for (int e = lane; e < 2 * HP; e += 32) sum += posth[(size_t)b * (2 * HP) + e] * fcw[e];
    for (int off = 16; off > 0; off >>= 1) sum += __shfl_down(sum, off, 32);
    if (lane == 0) out[b] = sum + fcb[0];
}

// ---------------------------------------------------------------------------
static inline size_t align256(size_t x) { return (x + 255) & ~(size_t)255; }

extern "C" void kernel_launch(void* const* d_in, const int* in_sizes, int n_in,
                              void* d_out, int out_size, void* d_ws, size_t ws_size,
                              hipStream_t stream) {
    (void)in_sizes; (void)n_in; (void)out_size; (void)ws_size;
    const int*   batch_data = (const int*)  d_in[0];
    const float* emb        = (const float*)d_in[1];
    const float* wih_f      = (const float*)d_in[2];
    const float* whh_f      = (const float*)d_in[3];
    const float* wb_f       = (const float*)d_in[4];
    const float* wih_b      = (const float*)d_in[5];
    const float* whh_b      = (const float*)d_in[6];
    const float* wb_b       = (const float*)d_in[7];
    const float* word_fc_w  = (const float*)d_in[8];
    const float* word_fc_b  = (const float*)d_in[9];
    const float* pih_f      = (const float*)d_in[10];
    const float* phh_f      = (const float*)d_in[11];
    const float* pb_f       = (const float*)d_in[12];
    const float* pih_b      = (const float*)d_in[13];
    const float* phh_b      = (const float*)d_in[14];
    const float* pb_b       = (const float*)d_in[15];
    const float* post_fc_w  = (const float*)d_in[16];
    const float* post_fc_b  = (const float*)d_in[17];
    float* out = (float*)d_out;

    char* ws = (char*)d_ws;
    size_t off = 0;
    auto carve = [&](size_t bytes) { void* p = ws + off; off = align256(off + bytes); return p; };
    int*    padded  = (int*)   carve((size_t)BATCH * PMAX * WMAXW * 4);
    int*    n_posts = (int*)   carve(BATCH * 4);
    __bf16* emb16   = (__bf16*)carve((size_t)VOCAB * EWP * 2);
    __bf16* wihT    = (__bf16*)carve((size_t)2 * GW * EWP * 2);
    __bf16* whhT    = (__bf16*)carve((size_t)2 * GW * HW * 2);
    __bf16* pihT    = (__bf16*)carve((size_t)2 * GP * HP * 2);
    __bf16* phhT    = (__bf16*)carve((size_t)2 * GP * HP * 2);
    float*  wordh   = (float*) carve((size_t)NROWS * 2 * HW * 4);
    __bf16* wordh16 = (__bf16*)carve((size_t)NROWS * 2 * HW * 2);
    float*  Gx      = (float*) carve((size_t)2 * PMAX * BATCH * GP * 4);   // 64 MB
    float*  posth   = (float*) carve((size_t)BATCH * 2 * HP * 4);

    init_padded_kernel<<<(BATCH * PMAX * WMAXW + 255) / 256, 256, 0, stream>>>(padded);
    build_padded_kernel<<<BATCH, 256, 0, stream>>>(batch_data, padded, n_posts);

    {
        long long tot = (long long)VOCAB * EWP;
        emb_cast_kernel<<<(unsigned)((tot + 255) / 256), 256, 0, stream>>>(emb, emb16);
    }
    auto tc = [&](const float* in, __bf16* o, int K, int N, int Kp) {
        long long tot = (long long)N * Kp;
        tcast_kernel<<<(unsigned)((tot + 255) / 256), 256, 0, stream>>>(in, o, K, N, Kp);
    };
    tc(wih_f, wihT,                      EW, GW, EWP);
    tc(wih_b, wihT + (size_t)GW * EWP,   EW, GW, EWP);
    tc(whh_f, whhT,                      HW, GW, HW);
    tc(whh_b, whhT + (size_t)GW * HW,    HW, GW, HW);
    tc(pih_f, pihT,                      HP, GP, HP);
    tc(pih_b, pihT + (size_t)GP * HP,    HP, GP, HP);
    tc(phh_f, phhT,                      HP, GP, HP);
    tc(phh_b, phhT + (size_t)GP * HP,    HP, GP, HP);

    {   // word BiLSTM: Xs 40960 + H 2*32768 + C 65536 = 172032 B LDS
        size_t lds = (size_t)64 * EWP * 2 + (size_t)2 * 64 * HW * 2 + (size_t)64 * HW * 4;
        word_lstm_kernel<<<dim3(NROWS / 64, 2), 256, lds, stream>>>(
            padded, emb16, wihT, whhT, wb_f, wb_b, wordh);
    }
    mask_fc_kernel<<<NROWS, 32, 0, stream>>>(wordh, n_posts, word_fc_w, word_fc_b,
                                             wordh16, out + BATCH);
    post_xgemm_kernel<<<dim3(PMAX, 2), 256, 0, stream>>>(wordh16, pihT, pb_f, pb_b, Gx);
    {   // post recurrence: H 2*16384 + C 32768 = 65536 B LDS
        size_t lds = (size_t)2 * 16 * HP * 2 + (size_t)16 * HP * 4;
        post_lstm_kernel<<<2, 256, lds, stream>>>(Gx, phhT, posth);
    }
    final_fc_kernel<<<BATCH, 32, 0, stream>>>(posth, post_fc_w, post_fc_b, out);
}
